// PerformerAttention_48576080118322
// MI455X (gfx1250) — matrix operand order repaired and checked by reading, not compiled, not run on hardware
//
#include <hip/hip_runtime.h>

// ---------------------------------------------------------------------------
// Performer attention for MI455X (gfx1250, wave32, WMMA bf16 -> f32 accum).
//   cast x->bf16; pack w{q,k,v,o} into [K/32][N][32] WMMA-B layout
//   Q/K = x @ w.T  (GEMM, bf16 row-major out);  V = x @ wv.T (packed-B out)
//   KfT = phi(K)^T  (WMMA proj + cos/sin, bf16 [B,H,512,T])
//   KVext = Kf^T @ V  packed [16][80][32] per head; col 64 = K_sum (ksum)
//   ctx = (phi(Q) @ KVext) / max(z,1e-6)  (fused, Qf in LDS)
//   out = ctx @ wo.T  (f32)
// GEMM tile staging uses double-buffered global_load_async_to_lds_b128:
// async loads complete in order per wave, so s_wait_asynccnt 6 drains the
// previous tile while the just-issued tile is still in flight.
// ---------------------------------------------------------------------------

typedef __bf16 bf16;
typedef __attribute__((ext_vector_type(16))) __bf16 v16bf;
typedef __attribute__((ext_vector_type(8)))  __bf16 v8bf;
typedef __attribute__((ext_vector_type(8)))  float  v8f;

#define H_  16
#define T_  4096
#define D_  1024
#define DH  64
#define Mm  256     // random features m
#define F_  512     // 2m
#define BT  8192    // B*T

__device__ __forceinline__ bf16 f2bf(float f) {
  unsigned u = __builtin_bit_cast(unsigned, f);
  u += 0x7FFFu + ((u >> 16) & 1u);               // round-to-nearest-even
  unsigned short s = (unsigned short)(u >> 16);
  return __builtin_bit_cast(bf16, s);
}
__device__ __forceinline__ float bf2f(bf16 h) {
  unsigned short s = __builtin_bit_cast(unsigned short, h);
  unsigned u = ((unsigned)s) << 16;
  return __builtin_bit_cast(float, u);
}
__device__ __forceinline__ v16bf cat8(v8bf lo, v8bf hi) {
  return __builtin_shufflevector(lo, hi, 0,1,2,3,4,5,6,7,8,9,10,11,12,13,14,15);
}
__device__ __forceinline__ v8f wmma_bf16(v16bf a, v16bf b, v8f c) {
  return __builtin_amdgcn_wmma_f32_16x16x32_bf16(false, a, false, b, (short)0, c,
                                                 false, false);
}

// Async global -> LDS, 16 bytes per lane, tracked by ASYNCcnt (ISA 08, §4).
__device__ __forceinline__ void async_g2l_b128(unsigned lds_addr, const bf16* g) {
  asm volatile("global_load_async_to_lds_b128 %0, %1, off"
               :: "v"(lds_addr), "v"(g) : "memory");
}
__device__ __forceinline__ void wait_async0() {
  asm volatile("s_wait_asynccnt 0" ::: "memory");
}
// Keep the newest 6 async ops (one tile's worth) outstanding; drain the rest.
__device__ __forceinline__ void wait_async6() {
  asm volatile("s_wait_asynccnt 6" ::: "memory");
}
__device__ __forceinline__ unsigned lds_addr_of(const void* p) {
  return (unsigned)(unsigned long long)p;
}

// A fragment (16x32 bf16) from LDS tile [M][K] row-major, row stride in halves.
// Lane L -> M=L%16, K=(L/16)*8 + {0..7, 16..23}.
__device__ __forceinline__ v16bf ldsA(const bf16* base, int mt, int k0,
                                      int stride, int lane) {
  int m  = mt + (lane & 15);
  int kb = k0 + ((lane >> 4) << 3);
  const bf16* p = base + m * stride + kb;
  v8bf lo = *(const v8bf*)p;
  v8bf hi = *(const v8bf*)(p + 16);
  return cat8(lo, hi);
}
// B fragment (32x16 bf16) from LDS tile [N][K], lane L -> N=L%16, K=(L/16)*8 + {0..7, 16..23}.
__device__ __forceinline__ v16bf ldsB(const bf16* base, int nt, int k0,
                                      int stride, int lane) {
  int n  = nt + (lane & 15);
  int kb = k0 + ((lane >> 4) << 3);
  const bf16* p = base + n * stride + kb;
  v8bf lo = *(const v8bf*)p;
  v8bf hi = *(const v8bf*)(p + 16);
  return cat8(lo, hi);
}

// ---------------------------------------------------------------------------
// GEMM: C[M,N] = A[M,K] @ B[K,N], B pre-packed as [K/32][Np][32] (Np = full N).
// Block tile 128x64, BK=64, 256 threads = 8 waves (4x2, each wave 32x32).
// Double-buffered LDS; tile n+1 issued async before waiting on tile n.
// outMode: 0 = bf16 row-major (ldc), 1 = f32 row-major (ldc),
//          2 = V packed per head [bh][t/32][dh][t%32],
//          3 = KV packed rows    [m/32][n][m%32] (row width Np, + z*sCz).
// ---------------------------------------------------------------------------
__global__ __launch_bounds__(256)
void gemm_bf16_kernel(const bf16* __restrict__ A, const bf16* __restrict__ Bp,
                      void* __restrict__ C,
                      int M, int N, int K, int lda, int Np, int ldc,
                      int outMode, long sAz, long sBz, long sCz)
{
  __shared__ bf16 As[2][128 * 64];      // [M][K]
  __shared__ bf16 Bs[2][2 * 64 * 32];   // two k-chunks, each [N][K] = [64][32]
  const int tid  = threadIdx.x;
  const int lane = tid & 31;
  const int wave = tid >> 5;
  const int z    = blockIdx.z;
  const bf16* Ag = A  + (long)z * sAz;
  const bf16* Bg = Bp + (long)z * sBz;
  const int m0 = blockIdx.y * 128;
  const int n0 = blockIdx.x * 64;
  const int wm = (wave >> 1) * 32;
  const int wn = (wave & 1) * 32;

  // issue one BK=64 tile: 4 A-chunks + 2 B-chunks per thread = 6 async ops/wave
  auto issue = [&](int buf, int k0) {
    int r = tid >> 3, cg = (tid & 7) << 3;
    #pragma unroll
    for (int p = 0; p < 4; ++p) {
      int row = p * 32 + r;
      async_g2l_b128(lds_addr_of(&As[buf][row * 64 + cg]),
                     Ag + (long)(m0 + row) * lda + k0 + cg);
    }
    #pragma unroll
    for (int c = 0; c < 2; ++c) {
      const bf16* src = Bg + (long)((k0 >> 5) + c) * Np * 32 + n0 * 32 + tid * 8;
      async_g2l_b128(lds_addr_of(&Bs[buf][c * 2048 + tid * 8]), src);
    }
  };

  v8f acc[2][2] = {};

  issue(0, 0);
  for (int k0 = 0; k0 < K; k0 += 64) {
    const int cur = (k0 >> 6) & 1;
    const bool more = (k0 + 64) < K;
    if (more) { issue(cur ^ 1, k0 + 64); wait_async6(); }
    else      { wait_async0(); }
    __syncthreads();

    #pragma unroll
    for (int s = 0; s < 2; ++s) {
      const bf16* bs = &Bs[cur][s * 2048];
      v16bf af0 = ldsA(&As[cur][0], wm,      s * 32, 64, lane);
      v16bf af1 = ldsA(&As[cur][0], wm + 16, s * 32, 64, lane);
      v16bf bf0 = ldsB(bs, wn,      0, 32, lane);
      v16bf bf1 = ldsB(bs, wn + 16, 0, 32, lane);
      acc[0][0] = wmma_bf16(af0, bf0, acc[0][0]);
      acc[0][1] = wmma_bf16(af0, bf1, acc[0][1]);
      acc[1][0] = wmma_bf16(af1, bf0, acc[1][0]);
      acc[1][1] = wmma_bf16(af1, bf1, acc[1][1]);
    }
    __syncthreads();   // everyone done reading `cur` before it is overwritten
  }

  const long coff = (long)z * sCz;
  #pragma unroll
  for (int i = 0; i < 2; ++i) {
    int mb = m0 + wm + i * 16 + ((lane >> 4) << 3);
    #pragma unroll
    for (int j = 0; j < 2; ++j) {
      int n = n0 + wn + j * 16 + (lane & 15);
      #pragma unroll
      for (int r = 0; r < 8; ++r) {
        float v = acc[i][j][r];
        int m = mb + r;
        if (outMode == 0) {
          ((bf16*)C)[coff + (long)m * ldc + n] = f2bf(v);
        } else if (outMode == 1) {
          ((float*)C)[coff + (long)m * ldc + n] = v;
        } else if (outMode == 2) {         // V packed per head, k-dim = t
          int b = m >> 12, t = m & (T_ - 1);
          int h = n >> 6, dh = n & 63;
          long idx = ((((long)(b * H_ + h) * (T_ / 32) + (t >> 5)) * 64 + dh) << 5)
                     + (t & 31);
          ((bf16*)C)[idx] = f2bf(v);
        } else {                           // KV packed rows, k-dim = m
          long idx = coff + (long)((m >> 5) * Np + n) * 32 + (m & 31);
          ((bf16*)C)[idx] = f2bf(v);
        }
      }
    }
  }
}

// ---------------------------------------------------------------------------
// phi(K) transposed: per (b,h, 64-row tile) proj = K_head[64,64] @ rf^T,
// scale = exp(-|k|^2/2)/sqrt(m);  KfT[feat][t] = cos, KfT[feat+256][t] = sin.
// rf [256][64] row-major IS the [N][K] LDS layout. 256 threads = 8 waves.
// ---------------------------------------------------------------------------
__global__ __launch_bounds__(256)
void phi_k_kernel(const bf16* __restrict__ Kb, const bf16* __restrict__ rfb,
                  bf16* __restrict__ KfT)
{
  __shared__ bf16  As[64 * 64];
  __shared__ bf16  Bs[256 * 64];
  __shared__ float scl[64];
  const int tid  = threadIdx.x;
  const int lane = tid & 31;
  const int wave = tid >> 5;
  const int tt   = blockIdx.x & 63;
  const int bh   = blockIdx.x >> 6;
  const int b    = bh >> 4;
  const int h    = bh & 15;

  const bf16* Ag = Kb + ((long)b * T_ + tt * 64) * D_ + h * DH;
  {
    int r = tid >> 3, cg = (tid & 7) << 3;
    #pragma unroll
    for (int p = 0; p < 2; ++p) {
      int row = p * 32 + r;
      async_g2l_b128(lds_addr_of(&As[row * 64 + cg]), Ag + (long)row * D_ + cg);
    }
  }
  #pragma unroll
  for (int q = 0; q < 8; ++q)
    async_g2l_b128(lds_addr_of(&Bs[(q * 256 + tid) * 8]), rfb + (q * 256 + tid) * 8);
  wait_async0();
  __syncthreads();

  if (tid < 64) {
    float ns = 0.f;
    #pragma unroll 8
    for (int k = 0; k < 64; ++k) { float v = bf2f(As[tid * 64 + k]); ns += v * v; }
    scl[tid] = __expf(-0.5f * ns) * 0.0625f;   // 1/sqrt(256)
  }
  __syncthreads();

  v8f acc[4][2] = {};
  #pragma unroll
  for (int ks = 0; ks < 2; ++ks) {
    int k0 = ks * 32;
    v16bf bf0 = ldsB(Bs, wave * 32,      k0, 64, lane);
    v16bf bf1 = ldsB(Bs, wave * 32 + 16, k0, 64, lane);
    #pragma unroll
    for (int i = 0; i < 4; ++i) {
      v16bf af = ldsA(As, i * 16, k0, 64, lane);
      acc[i][0] = wmma_bf16(af, bf0, acc[i][0]);
      acc[i][1] = wmma_bf16(af, bf1, acc[i][1]);
    }
  }

  bf16* out = KfT + (long)bh * F_ * T_;
  #pragma unroll
  for (int i = 0; i < 4; ++i) {
    int tb = tt * 64 + i * 16 + ((lane >> 4) << 3);
    #pragma unroll
    for (int j = 0; j < 2; ++j) {
      int feat = wave * 32 + j * 16 + (lane & 15);
      #pragma unroll
      for (int r = 0; r < 8; ++r) {
        int t = tb + r;
        float p = acc[i][j][r];
        float s = scl[(t & 63)];
        out[(long)feat * T_ + t]        = f2bf(__cosf(p) * s);
        out[(long)(feat + Mm) * T_ + t] = f2bf(__sinf(p) * s);
      }
    }
  }
}

// ---------------------------------------------------------------------------
// Fused phi(Q) + QKV + normalize. KVe is packed [16][80][32] per head.
// Dynamic LDS: Qf 64KiB | region2 40KiB (phi: As+rf / qkv: Bs+Cout) | scl.
// ---------------------------------------------------------------------------
__global__ __launch_bounds__(256)
void phi_q_qkv_kernel(const bf16* __restrict__ Qb, const bf16* __restrict__ rfb,
                      const bf16* __restrict__ KVe, bf16* __restrict__ ctx)
{
  extern __shared__ char smem[];
  bf16*  Qf    = (bf16*)smem;                              // [64][512]
  bf16*  As    = (bf16*)(smem + 65536);                    // [64][64]   (phi)
  bf16*  Bs_rf = (bf16*)(smem + 65536 + 8192);             // [256][64]  (phi)
  bf16*  Bs    = (bf16*)(smem + 65536);                    // [80][32]   (qkv)
  float* Cout  = (float*)(smem + 65536 + 8192);            // [64][80]   (qkv)
  float* scl   = (float*)(smem + 65536 + 8192 + 32768);    // [64]

  const int tid  = threadIdx.x;
  const int lane = tid & 31;
  const int wave = tid >> 5;
  const int tt   = blockIdx.x & 63;
  const int bh   = blockIdx.x >> 6;
  const int b    = bh >> 4;
  const int h    = bh & 15;

  // ---- stage 1: phi(Q) into LDS ----
  const bf16* Ag = Qb + ((long)b * T_ + tt * 64) * D_ + h * DH;
  {
    int r = tid >> 3, cg = (tid & 7) << 3;
    #pragma unroll
    for (int p = 0; p < 2; ++p) {
      int row = p * 32 + r;
      async_g2l_b128(lds_addr_of(&As[row * 64 + cg]), Ag + (long)row * D_ + cg);
    }
  }
  #pragma unroll
  for (int q = 0; q < 8; ++q)
    async_g2l_b128(lds_addr_of(&Bs_rf[(q * 256 + tid) * 8]), rfb + (q * 256 + tid) * 8);
  wait_async0();
  __syncthreads();
  if (tid < 64) {
    float ns = 0.f;
    #pragma unroll 8
    for (int k = 0; k < 64; ++k) { float v = bf2f(As[tid * 64 + k]); ns += v * v; }
    scl[tid] = __expf(-0.5f * ns) * 0.0625f;
  }
  __syncthreads();

  {
    v8f acc[4][2] = {};
    #pragma unroll
    for (int ks = 0; ks < 2; ++ks) {
      int k0 = ks * 32;
      v16bf bf0 = ldsB(Bs_rf, wave * 32,      k0, 64, lane);
      v16bf bf1 = ldsB(Bs_rf, wave * 32 + 16, k0, 64, lane);
      #pragma unroll
      for (int i = 0; i < 4; ++i) {
        v16bf af = ldsA(As, i * 16, k0, 64, lane);
        acc[i][0] = wmma_bf16(af, bf0, acc[i][0]);
        acc[i][1] = wmma_bf16(af, bf1, acc[i][1]);
      }
    }
    #pragma unroll
    for (int i = 0; i < 4; ++i) {
      int tb = i * 16 + ((lane >> 4) << 3);
      #pragma unroll
      for (int j = 0; j < 2; ++j) {
        int feat = wave * 32 + j * 16 + (lane & 15);
        #pragma unroll
        for (int r = 0; r < 8; ++r) {
          int t = tb + r;
          float p = acc[i][j][r];
          float s = scl[t];
          Qf[t * F_ + feat]      = f2bf(__cosf(p) * s);
          Qf[t * F_ + feat + Mm] = f2bf(__sinf(p) * s);
        }
      }
    }
  }
  __syncthreads();   // Qf ready; region2 reusable

  // ---- stage 2: C[64,80] = Qf @ KVext (packed B) ----
  const bf16* Bgp = KVe + (long)bh * F_ * 80;
  v8f acc2[4] = {};
  for (int k0 = 0; k0 < F_; k0 += 32) {
    const bf16* src = Bgp + (long)(k0 >> 5) * 2560;
    for (int e = tid; e < 320; e += 256)
      async_g2l_b128(lds_addr_of(&Bs[e * 8]), src + e * 8);
    wait_async0();
    __syncthreads();
    if (wave < 5) {
      v16bf bfr = ldsB(Bs, wave * 16, 0, 32, lane);
      #pragma unroll
      for (int i = 0; i < 4; ++i) {
        v16bf af = ldsA(Qf, i * 16, k0, 512, lane);
        acc2[i] = wmma_bf16(af, bfr, acc2[i]);
      }
    }
    __syncthreads();
  }
  if (wave < 5) {
    #pragma unroll
    for (int i = 0; i < 4; ++i) {
      int tb = i * 16 + ((lane >> 4) << 3);
      int n  = wave * 16 + (lane & 15);
      #pragma unroll
      for (int r = 0; r < 8; ++r) Cout[(tb + r) * 80 + n] = acc2[i][r];
    }
  }
  __syncthreads();

  // ---- stage 3: normalize, scatter to ctx ----
  bf16* outp = ctx + ((long)b * T_ + tt * 64) * D_ + h * DH;
  for (int e = tid; e < 4096; e += 256) {
    int t = e >> 6, dd = e & 63;
    float zv = fmaxf(Cout[t * 80 + 64], 1e-6f);
    outp[(long)t * D_ + dd] = f2bf(Cout[t * 80 + dd] / zv);
  }
}

// K_sum: wave per KfT row (coalesced), shuffle-tree reduce; writes packed
// KVe col 64 (and zeros cols 65..79 at this row's kk slot).
__global__ __launch_bounds__(256)
void ksum_kernel(const bf16* __restrict__ KfT, bf16* __restrict__ KVe)
{
  const int lane = threadIdx.x & 31;
  const int wave = threadIdx.x >> 5;
  int gf = blockIdx.x * 8 + wave;          // 0 .. 32*512-1
  int bh = gf >> 9, f = gf & (F_ - 1);
  const bf16* row = KfT + ((long)bh * F_ + f) * T_;
  float s = 0.f;
  for (int t = lane; t < T_; t += 32) s += bf2f(row[t]);
  #pragma unroll
  for (int off = 16; off > 0; off >>= 1) s += __shfl_down(s, off, 32);
  bf16* chunk = KVe + (long)bh * F_ * 80 + (long)(f >> 5) * 2560 + (f & 31);
  if (lane == 0)            chunk[64 * 32]          = f2bf(s);
  else if (lane < 16)       chunk[(64 + lane) * 32] = f2bf(0.f);
}

__global__ void cast_f32_bf16_kernel(const float* __restrict__ in,
                                     bf16* __restrict__ out, long n)
{
  long i = (long)blockIdx.x * blockDim.x + threadIdx.x;
  if (i < n) out[i] = f2bf(in[i]);
}

// W[1024][1024] f32 -> packed Wp[k/32][n][k%32] = bf16(W[n][k])
__global__ void pack_w_kernel(const float* __restrict__ W, bf16* __restrict__ Wp)
{
  long id = (long)blockIdx.x * 256 + threadIdx.x;
  if (id < (long)D_ * D_) {
    int kt  = (int)(id >> 15);           // 1024*32 elems per chunk
    int rem = (int)(id & 32767);
    int n = rem >> 5, kk = rem & 31;
    Wp[id] = f2bf(W[(long)n * D_ + kt * 32 + kk]);
  }
}

// ---------------------------------------------------------------------------
extern "C" void kernel_launch(void* const* d_in, const int* in_sizes, int n_in,
                              void* d_out, int out_size, void* d_ws, size_t ws_size,
                              hipStream_t stream)
{
  const float* x  = (const float*)d_in[0];
  const float* wq = (const float*)d_in[1];
  const float* wk = (const float*)d_in[2];
  const float* wv = (const float*)d_in[3];
  const float* wo = (const float*)d_in[4];
  const float* rf = (const float*)d_in[5];
  float* out = (float*)d_out;

  char* p = (char*)d_ws;
  auto carve = [&](size_t bytes) {
    char* q = p; p += (bytes + 255) & ~(size_t)255; return q;
  };
  bf16* xb  = (bf16*)carve((size_t)BT * D_ * 2);
  bf16* Qb  = (bf16*)carve((size_t)BT * D_ * 2);
  bf16* Kb  = (bf16*)carve((size_t)BT * D_ * 2);
  bf16* Vp  = (bf16*)carve((size_t)BT * D_ * 2);   // packed per head
  bf16* ctx = (bf16*)carve((size_t)BT * D_ * 2);
  bf16* wqP = (bf16*)carve((size_t)D_ * D_ * 2);
  bf16* wkP = (bf16*)carve((size_t)D_ * D_ * 2);
  bf16* wvP = (bf16*)carve((size_t)D_ * D_ * 2);
  bf16* woP = (bf16*)carve((size_t)D_ * D_ * 2);
  bf16* rfb = (bf16*)carve((size_t)Mm * DH * 2);
  bf16* KfT = (bf16*)carve((size_t)32 * F_ * T_ * 2);   // 128 MiB
  bf16* KVe = (bf16*)carve((size_t)32 * F_ * 80 * 2);   // packed

  cast_f32_bf16_kernel<<<(BT * D_ + 255) / 256, 256, 0, stream>>>(x, xb, (long)BT * D_);
  cast_f32_bf16_kernel<<<(Mm * DH + 255) / 256, 256, 0, stream>>>(rf, rfb, (long)Mm * DH);
  pack_w_kernel<<<4096, 256, 0, stream>>>(wq, wqP);
  pack_w_kernel<<<4096, 256, 0, stream>>>(wk, wkP);
  pack_w_kernel<<<4096, 256, 0, stream>>>(wv, wvP);
  pack_w_kernel<<<4096, 256, 0, stream>>>(wo, woP);

  // Q/K/V projections: [8192,1024] = xb @ w.T
  dim3 gproj(D_ / 64, BT / 128, 1);
  gemm_bf16_kernel<<<gproj, 256, 0, stream>>>(xb, wqP, Qb, BT, D_, D_, D_, D_, D_,
                                              0, 0, 0, 0);
  gemm_bf16_kernel<<<gproj, 256, 0, stream>>>(xb, wkP, Kb, BT, D_, D_, D_, D_, D_,
                                              0, 0, 0, 0);
  gemm_bf16_kernel<<<gproj, 256, 0, stream>>>(xb, wvP, Vp, BT, D_, D_, D_, D_, D_,
                                              2, 0, 0, 0);

  // KfT = phi(K)^T
  phi_k_kernel<<<32 * (T_ / 64), 256, 0, stream>>>(Kb, rfb, KfT);

  // KVext[:, :64] = Kf^T @ V per head (z = b*16+h), packed output
  dim3 gkv(1, F_ / 128, 32);
  gemm_bf16_kernel<<<gkv, 256, 0, stream>>>(KfT, Vp, KVe,
                                            F_, DH, T_, T_, DH, 0,
                                            3,
                                            (long)F_ * T_,   // A per head
                                            (long)T_ * DH,   // packed V per head
                                            (long)F_ * 80);  // packed C per head

  // KVext col 64 = K_sum, cols 65..79 = 0
  ksum_kernel<<<(32 * F_) / 8, 256, 0, stream>>>(KfT, KVe);

  // fused phi(Q) + QKV + normalize -> ctx
  size_t shq = 65536 + 8192 + 32768 + 256;
  phi_q_qkv_kernel<<<32 * (T_ / 64), 256, shq, stream>>>(Qb, rfb, KVe, ctx);

  // out = ctx @ wo.T  (f32)
  gemm_bf16_kernel<<<gproj, 256, 0, stream>>>(ctx, woP, out, BT, D_, D_, D_, D_, D_,
                                              1, 0, 0, 0);
}
